// gnn_encoder_82592221102364
// MI455X (gfx1250) — matrix operand order repaired and checked by reading, not compiled, not run on hardware
//
#include <hip/hip_runtime.h>
#include <hip/hip_bf16.h>
#include <math.h>

// ---------------------------------------------------------------------------
// GGNN encoder for MI455X (gfx1250, wave32, WMMA).
// Heavy op: act[e] = edges[e]^T @ h  (M=3072, K=3072, N=32) x4 edge types x15.
// Strategy: edges (0/1, exact in bf16) pre-swizzled ONCE into WMMA A-fragment
// layout (75.5 MB, L2-resident); h packed to B-fragment layout each step;
// aggregation runs V_WMMA_F32_16X16X32_BF16 with f32 accumulators and the
// K dimension split 8 ways (6144 waves) to fill all WGPs / saturate L2 BW;
// a deterministic reduce sums partials. Gating is f32 VALU (tiny).
// Workspace: ~90 MB.
// ---------------------------------------------------------------------------

typedef __attribute__((ext_vector_type(16))) __bf16 v16bf;
typedef __attribute__((ext_vector_type(8)))  float  v8f;

#define NN   3072
#define FIN  64
#define EE   4
#define SS   32
#define LLAY 3
#define GITR 5
#define NT   (NN / 16)       // 192 n-tiles (output rows)
#define KT   (NN / 32)       // 96  k-tiles (reduction)
#define KSPLIT 8             // K-split factor for occupancy
#define KTW  (KT / KSPLIT)   // 12 k-tiles per wave

__device__ __forceinline__ unsigned bf16rne(float f) {
  unsigned u = __float_as_uint(f);
  u += 0x7FFFu + ((u >> 16) & 1u);     // round-to-nearest-even into bf16
  return u >> 16;
}
__device__ __forceinline__ unsigned bf16pair(float lo, float hi) {
  return bf16rne(lo) | (bf16rne(hi) << 16);
}

// ---------------- h = x @ fc_w + fc_b  ([3072,64]@[64,32]) ------------------
__global__ __launch_bounds__(256) void fc_kernel(const float* __restrict__ x,
                                                 const float* __restrict__ w,
                                                 const float* __restrict__ b,
                                                 float* __restrict__ hF) {
  int idx = blockIdx.x * 256 + threadIdx.x;      // 98304 outputs
  int n = idx >> 5, s = idx & 31;
  float acc = b[s];
  const float* xr = x + (size_t)n * FIN;
#pragma unroll 8
  for (int k = 0; k < FIN; ++k) acc += xr[k] * w[k * SS + s];
  hF[idx] = acc;
}

// ------- pack h (f32 row-major) into bf16 WMMA B-fragment layout ------------
// hB[(kt*2+st)*256 + lane*8 + v] holds h[kt*32 + (v<4 ? 2v : 16+2(v-4)) + (lane>>4)*8 (+1)][st*16 + lane&15]
__global__ __launch_bounds__(256) void pack_hB_kernel(const float* __restrict__ hF,
                                                      unsigned* __restrict__ hB) {
  int o = blockIdx.x * 256 + threadIdx.x;        // 49152 dwords
  int v    = o & 7;
  int lane = (o >> 3) & 31;
  int st   = (o >> 8) & 1;
  int kt   = o >> 9;
  int m = kt * 32 + ((v < 4) ? (2 * v) : (16 + 2 * (v - 4))) + (lane >> 4) * 8;
  int s = st * 16 + (lane & 15);
  hB[o] = bf16pair(hF[(size_t)m * SS + s], hF[(size_t)(m + 1) * SS + s]);
}

// ---- pre-swizzle edges (f32 0/1) into bf16 WMMA A-fragment layout, once ----
// A tile = edgesT[e][n0:n0+16][m0:m0+32]; per-lane K packing per CDNA5 ISA:
//   lane half=lane>>4, row n'=lane&15; v<4 -> K=2v+8*half, v>=4 -> K=16+2(v-4)+8*half
__global__ __launch_bounds__(256) void prep_edges_kernel(const float* __restrict__ edges,
                                                         unsigned* __restrict__ eA) {
  const int wid  = threadIdx.x >> 5;
  const int lane = threadIdx.x & 31;
  const int tile = blockIdx.x * 8 + wid;         // [0, EE*NT*KT) = 73728
  const int kt   = tile % KT;
  const int ent  = tile / KT;                    // e*NT + nt
  const int half = lane >> 4, nn = lane & 15;
  const int n = (ent % NT) * 16 + nn;
  const float* eb = edges + (size_t)(ent / NT) * NN * NN + n;
  unsigned* ob = eA + (size_t)tile * 256 + lane * 8;
#pragma unroll
  for (int v = 0; v < 8; ++v) {
    int kk = (v < 4) ? (2 * v + half * 8) : (16 + 2 * (v - 4) + half * 8);
    int m = kt * 32 + kk;
    ob[v] = bf16pair(eb[(size_t)m * NN], eb[(size_t)(m + 1) * NN]);  // A[n', K], A[n', K+1]
  }
}

// -------- heavy: partial act = edges^T @ h via WMMA, K split 8 ways ---------
// job = ks*(EE*NT) + (e*NT + nt): the 8 waves of a block share one ks so they
// reuse the same hB cache lines; 768 blocks fill the machine.
__global__ __launch_bounds__(256) void agg_kernel(const unsigned* __restrict__ eA,
                                                  const unsigned* __restrict__ hB,
                                                  float* __restrict__ actP) {
  const int wid  = threadIdx.x >> 5;
  const int lane = threadIdx.x & 31;
  const int job  = blockIdx.x * 8 + wid;         // [0, KSPLIT*EE*NT) = 6144 waves
  const int ks   = job / (EE * NT);
  const int ent  = job % (EE * NT);              // e*NT + nt
  v8f c0 = {};                                   // s = 0..15
  v8f c1 = {};                                   // s = 16..31
  const unsigned* ap = eA + ((size_t)ent * KT + ks * KTW) * 256 + lane * 8;
  const unsigned* bp = hB + (size_t)(ks * KTW) * 512 + lane * 8;
#pragma unroll
  for (int kt = 0; kt < KTW; ++kt) {
    v16bf a  = *(const v16bf*)(ap);
    v16bf b0 = *(const v16bf*)(bp);
    v16bf b1 = *(const v16bf*)(bp + 256);
    c0 = __builtin_amdgcn_wmma_f32_16x16x32_bf16(false, a, false, b0, (short)0, c0, false, false);
    c1 = __builtin_amdgcn_wmma_f32_16x16x32_bf16(false, a, false, b1, (short)0, c1, false, false);
    ap += 256;
    bp += 512;
  }
  // C/D layout: VGPR j -> row j (lanes 0-15) / row 8+j (lanes 16-31), col = lane&15
  const int half = lane >> 4, nn = lane & 15;
  const int nt = ent % NT;
  float* o = actP + ((size_t)ks * EE * NN + (size_t)(ent / NT) * NN + nt * 16 + half * 8) * SS + nn;
#pragma unroll
  for (int j = 0; j < 8; ++j) {
    o[(size_t)j * SS]      = c0[j];
    o[(size_t)j * SS + 16] = c1[j];
  }
}

// ----------- deterministic reduction of the 8 K-split partials --------------
__global__ __launch_bounds__(256) void reduce_kernel(const float* __restrict__ actP,
                                                     float* __restrict__ act) {
  int idx = blockIdx.x * 256 + threadIdx.x;      // EE*NN*SS = 393216 outputs
  float s = 0.f;
#pragma unroll
  for (int ks = 0; ks < KSPLIT; ++ks)
    s += actP[(size_t)ks * (EE * NN * SS) + idx];
  act[idx] = s;
}

// ---------------- GRU-style gating (one wave per node, f32) -----------------
__global__ __launch_bounds__(256) void gate_kernel(const float* __restrict__ act,
                                                   float* __restrict__ hF,
                                                   const float* __restrict__ wz,
                                                   const float* __restrict__ wr,
                                                   const float* __restrict__ wh,
                                                   const float* __restrict__ uz,
                                                   const float* __restrict__ ur,
                                                   const float* __restrict__ uh,
                                                   const float* __restrict__ bz,
                                                   const float* __restrict__ br,
                                                   int l) {
  __shared__ float rh[8][SS];
  const int wid  = threadIdx.x >> 5;
  const int s    = threadIdx.x & 31;
  const int node = blockIdx.x * 8 + wid;
  const float* hrow = hF + (size_t)node * SS;

  float az = bz[l * SS + s];
  float ar = br[l * SS + s];
  float ah = 0.f;
  for (int e = 0; e < EE; ++e) {
    const float* arow = act + ((size_t)e * NN + node) * SS;
    const float* wzb = wz + ((size_t)(l * EE + e) * SS) * SS + s;
    const float* wrb = wr + ((size_t)(l * EE + e) * SS) * SS + s;
    const float* whb = wh + ((size_t)(l * EE + e) * SS) * SS + s;
#pragma unroll
    for (int k = 0; k < SS; ++k) {
      float a = arow[k];
      az += a * wzb[(size_t)k * SS];
      ar += a * wrb[(size_t)k * SS];
      ah += a * whb[(size_t)k * SS];
    }
  }
  const float* uzb = uz + (size_t)l * SS * SS + s;
  const float* urb = ur + (size_t)l * SS * SS + s;
#pragma unroll
  for (int k = 0; k < SS; ++k) {
    float hv = hrow[k];
    az += hv * uzb[(size_t)k * SS];
    ar += hv * urb[(size_t)k * SS];
  }
  float z  = 1.f / (1.f + expf(-az));
  float r  = 1.f / (1.f + expf(-ar));
  float hs = hrow[s];
  rh[wid][s] = r * hs;
  __syncthreads();
  const float* uhb = uh + (size_t)l * SS * SS + s;
#pragma unroll
  for (int k = 0; k < SS; ++k) ah += rh[wid][k] * uhb[(size_t)k * SS];
  float hh = tanhf(ah);
  hF[(size_t)node * SS + s] = z * hs + (1.f - z) * hh;
}

// ---------------- y = log_softmax(h[5] @ out_w + out_b) ---------------------
__global__ void out_kernel(const float* __restrict__ hF,
                           const float* __restrict__ ow,
                           const float* __restrict__ ob,
                           float* __restrict__ out) {
  if (threadIdx.x == 0) {
    float y[5];
#pragma unroll
    for (int j = 0; j < 5; ++j) y[j] = ob[j];
    for (int s = 0; s < SS; ++s) {
      float hv = hF[5 * SS + s];
#pragma unroll
      for (int j = 0; j < 5; ++j) y[j] += hv * ow[s * 5 + j];
    }
    float m = y[0];
#pragma unroll
    for (int j = 1; j < 5; ++j) m = fmaxf(m, y[j]);
    float sum = 0.f;
#pragma unroll
    for (int j = 0; j < 5; ++j) sum += expf(y[j] - m);
    float ls = m + logf(sum);
#pragma unroll
    for (int j = 0; j < 5; ++j) out[j] = y[j] - ls;
  }
}

extern "C" void kernel_launch(void* const* d_in, const int* in_sizes, int n_in,
                              void* d_out, int out_size, void* d_ws, size_t ws_size,
                              hipStream_t stream) {
  const float* x     = (const float*)d_in[0];
  // d_in[1] = x_lengths (unused: all nodes valid)
  const float* edges = (const float*)d_in[2];
  const float* fc_w  = (const float*)d_in[3];
  const float* fc_b  = (const float*)d_in[4];
  const float* wz    = (const float*)d_in[5];
  const float* wr    = (const float*)d_in[6];
  const float* wh    = (const float*)d_in[7];
  const float* uz    = (const float*)d_in[8];
  const float* ur    = (const float*)d_in[9];
  const float* uh    = (const float*)d_in[10];
  const float* bz    = (const float*)d_in[11];
  const float* br    = (const float*)d_in[12];
  const float* ow    = (const float*)d_in[13];
  const float* obv   = (const float*)d_in[14];
  float* out = (float*)d_out;

  // workspace carve-up (all 32B aligned): ~90 MB total
  float*    hF   = (float*)d_ws;                              // 98304 f32  (h, row-major)
  float*    act  = hF + (size_t)NN * SS;                      // 393216 f32 (summed agg)
  float*    actP = act + (size_t)EE * NN * SS;                // 8*393216 f32 (K-split partials)
  unsigned* hB   = (unsigned*)(actP + (size_t)KSPLIT * EE * NN * SS); // 49152 u32 (h B-frags)
  unsigned* eA   = hB + (size_t)KT * 2 * 256;                 // 18874368 u32 (edge A-frags)

  fc_kernel<<<384, 256, 0, stream>>>(x, fc_w, fc_b, hF);
  pack_hB_kernel<<<192, 256, 0, stream>>>(hF, hB);
  prep_edges_kernel<<<9216, 256, 0, stream>>>(edges, eA);

  for (int l = 0; l < LLAY; ++l) {
    for (int it = 0; it < GITR; ++it) {
      agg_kernel<<<768, 256, 0, stream>>>(eA, hB, actP);
      reduce_kernel<<<1536, 256, 0, stream>>>(actP, act);
      gate_kernel<<<384, 256, 0, stream>>>(act, hF, wz, wr, wh, uz, ur, uh, bz, br, l);
      pack_hB_kernel<<<192, 256, 0, stream>>>(hF, hB);
    }
  }
  out_kernel<<<1, 32, 0, stream>>>(hF, ow, obv, out);
}
